// MultiHeadSelfAttention_13391708029843
// MI455X (gfx1250) — matrix operand order repaired and checked by reading, not compiled, not run on hardware
//
#include <hip/hip_runtime.h>

#define DEV __device__ __forceinline__

typedef __attribute__((ext_vector_type(4)))  unsigned short u16x4;
typedef __attribute__((ext_vector_type(8)))  unsigned short u16x8;
typedef __attribute__((ext_vector_type(16))) unsigned short u16x16;
typedef __attribute__((ext_vector_type(16))) __bf16         bf16x16;
typedef __attribute__((ext_vector_type(8)))  float          f32x8;

static DEV unsigned short f32_to_bf16(float f) {
  union { float f; unsigned u; } v; v.f = f;
  unsigned r = v.u + 0x7fffu + ((v.u >> 16) & 1u);   // round-to-nearest-even
  return (unsigned short)(r >> 16);
}

union FragU { u16x16 u; bf16x16 v; };

static DEV bf16x16 make_frag(u16x8 lo, u16x8 hi) {
  FragU f;
#pragma unroll
  for (int i = 0; i < 8; ++i) { f.u[i] = lo[i]; f.u[i + 8] = hi[i]; }
  return f.v;
}

static DEV f32x8 wmma_bf16(bf16x16 a, bf16x16 b, f32x8 c) {
  return __builtin_amdgcn_wmma_f32_16x16x32_bf16(false, a, false, b, (short)0, c,
                                                 false, false);
}

// LDS byte offset of a generic pointer to __shared__ data (ISA: LDS aperture
// generic addresses carry the LDS offset in addr[31:0]).
static DEV unsigned lds_offset(const void* p) {
  return (unsigned)(uintptr_t)p;
}

// Async global->LDS copy, 16 bytes per lane (GLOBAL_LOAD_ASYNC_TO_LDS_B128,
// tracked by ASYNCcnt). Bypasses VGPRs entirely.
static DEV void async_copy_b128(unsigned lds_off, const void* gptr) {
  asm volatile("global_load_async_to_lds_b128 %0, %1, off"
               :: "v"(lds_off), "v"((unsigned long long)(uintptr_t)gptr)
               : "memory");
}

static DEV void wait_asynccnt0() {
  asm volatile("s_wait_asynccnt 0x0" ::: "memory");
}

// ---------------------------------------------------------------------------
// Tiled GEMM:  C[M,N] = A[M,K] (f32) @ W[N,K]^T (f32) + bias[N]
// Block tile 128x128, K-step 32 (bf16 WMMA), 256 threads = 8 waves (4M x 2N),
// double-buffered LDS (one barrier per K-step, global loads overlapped).
// MODE 0: scatter bf16 into QKV workspace laid out (s, b, h, t, hd);
//         Q (s==0) is pre-scaled by 1/sqrt(HD) = 0.125 (exact in bf16).
// MODE 1: write f32 to out[M,N].
// ---------------------------------------------------------------------------
template <int MODE>
__global__ __launch_bounds__(256) void gemm_xwT(
    const float* __restrict__ A, const float* __restrict__ W,
    const float* __restrict__ bias,
    unsigned short* __restrict__ qkv_out, float* __restrict__ f32_out,
    int M, int N, int K)
{
  __shared__ unsigned short lsA[2][128 * 40];   // 128 rows x 32 bf16, stride 40
  __shared__ unsigned short lsB[2][128 * 40];

  const int tid  = threadIdx.x;
  const int wv   = tid >> 5;
  const int lane = tid & 31;
  const int half = lane >> 4;
  const int l16  = lane & 15;
  const int waveM = wv >> 1;   // 0..3
  const int waveN = wv & 1;    // 0..1
  const int m0 = blockIdx.y * 128;
  const int n0 = blockIdx.x * 128;

  float4 ra[4], rb[4];

  auto load_tiles = [&](int k0) {
#pragma unroll
    for (int p = 0; p < 4; ++p) {
      int idx = tid + p * 256;
      int r = idx >> 3, c4 = (idx & 7) * 4;
      ra[p] = *(const float4*)(A + (size_t)(m0 + r) * K + k0 + c4);
      rb[p] = *(const float4*)(W + (size_t)(n0 + r) * K + k0 + c4);
    }
  };
  auto store_tiles = [&](int buf) {
#pragma unroll
    for (int p = 0; p < 4; ++p) {
      int idx = tid + p * 256;
      int r = idx >> 3, c4 = (idx & 7) * 4;
      u16x4 ap;
      ap[0] = f32_to_bf16(ra[p].x); ap[1] = f32_to_bf16(ra[p].y);
      ap[2] = f32_to_bf16(ra[p].z); ap[3] = f32_to_bf16(ra[p].w);
      *(u16x4*)&lsA[buf][r * 40 + c4] = ap;
      u16x4 bp;
      bp[0] = f32_to_bf16(rb[p].x); bp[1] = f32_to_bf16(rb[p].y);
      bp[2] = f32_to_bf16(rb[p].z); bp[3] = f32_to_bf16(rb[p].w);
      *(u16x4*)&lsB[buf][r * 40 + c4] = bp;
    }
  };

  f32x8 acc[2][4];
#pragma unroll
  for (int i = 0; i < 2; ++i)
#pragma unroll
    for (int j = 0; j < 4; ++j)
#pragma unroll
      for (int e = 0; e < 8; ++e) acc[i][j][e] = 0.f;

  load_tiles(0);
  store_tiles(0);
  __syncthreads();

  int pb = 0;
  for (int k0 = 0; k0 < K; k0 += 32) {
    const bool haveNext = (k0 + 32 < K);
    if (haveNext) load_tiles(k0 + 32);   // overlap with WMMAs below

    bf16x16 af[2];
#pragma unroll
    for (int mt = 0; mt < 2; ++mt) {
      int row = waveM * 32 + mt * 16 + l16;
      u16x8 lo = *(const u16x8*)&lsA[pb][row * 40 + half * 8];
      u16x8 hi = *(const u16x8*)&lsA[pb][row * 40 + 16 + half * 8];
      af[mt] = make_frag(lo, hi);
    }
    bf16x16 bfr[4];
#pragma unroll
    for (int nt = 0; nt < 4; ++nt) {
      int row = waveN * 64 + nt * 16 + l16;
      u16x8 lo = *(const u16x8*)&lsB[pb][row * 40 + half * 8];
      u16x8 hi = *(const u16x8*)&lsB[pb][row * 40 + 16 + half * 8];
      bfr[nt] = make_frag(lo, hi);
    }
#pragma unroll
    for (int mt = 0; mt < 2; ++mt)
#pragma unroll
      for (int nt = 0; nt < 4; ++nt)
        acc[mt][nt] = wmma_bf16(af[mt], bfr[nt], acc[mt][nt]);

    if (haveNext) {
      store_tiles(pb ^ 1);
      __syncthreads();
      pb ^= 1;
    }
  }

  // --- epilogue ---
#pragma unroll
  for (int mt = 0; mt < 2; ++mt)
#pragma unroll
    for (int nt = 0; nt < 4; ++nt) {
      int gn = n0 + waveN * 64 + nt * 16 + l16;
      float bb = bias[gn];
#pragma unroll
      for (int j = 0; j < 8; ++j) {
        int gm = m0 + waveM * 32 + mt * 16 + half * 8 + j;
        float v = acc[mt][nt][j] + bb;
        if (MODE == 1) {
          f32_out[(size_t)gm * N + gn] = v;
        } else {
          int s = gn >> 10, c = gn & 1023;
          int h = c >> 6,  d = c & 63;
          int b = gm >> 11, t = gm & 2047;
          if (s == 0) v *= 0.125f;   // fold softmax scale into Q (exact: 2^-3)
          qkv_out[((((size_t)s * 4 + b) * 16 + h) * 2048 + t) * 64 + d] =
              f32_to_bf16(v);
        }
      }
    }
}

// ---------------------------------------------------------------------------
// Flash attention: grid = (T/128, B*H). Block = 256 threads = 8 waves,
// each wave owns 16 Q rows of a 128-row Q tile. K tiles stream via
// GLOBAL_LOAD_ASYNC_TO_LDS_B128 into double-buffered LDS; V tiles are
// transposed through registers into double-buffered LDS; online softmax
// (Q pre-scaled, so raw WMMA scores are used directly); P round-trips
// per-wave LDS (C-layout -> A-layout) for the P@V WMMAs.
// ---------------------------------------------------------------------------
__global__ __launch_bounds__(256) void flash_attn(
    const unsigned short* __restrict__ qkv, float* __restrict__ attn_out)
{
  __shared__ unsigned short lsK[2][128 * 72];     // K tile, row stride 72
  __shared__ unsigned short lsV[2][64 * 136];     // V^T tile, row stride 136
  __shared__ unsigned short lsP[8 * 16 * 136];    // per-wave P (16x128)

  const int tid  = threadIdx.x;
  const int wv   = tid >> 5;
  const int lane = tid & 31;
  const int half = lane >> 4;
  const int l16  = lane & 15;

  const int bh = blockIdx.y;                 // b*16 + h
  const int m0 = blockIdx.x * 128;
  const size_t plane = (size_t)2048 * 64;
  const unsigned short* qb = qkv + (size_t)bh * plane;
  const unsigned short* kb = qkv + (size_t)(64 + bh) * plane;
  const unsigned short* vb = qkv + (size_t)(128 + bh) * plane;

  u16x8 vreg[4];

  auto issue_k_async = [&](int buf, int k0g) {
#pragma unroll
    for (int p = 0; p < 4; ++p) {
      int idx = tid + p * 256;
      int r = idx >> 3, c8 = (idx & 7) * 8;
      async_copy_b128(lds_offset(&lsK[buf][r * 72 + c8]),
                      kb + (size_t)(k0g + r) * 64 + c8);
    }
  };
  auto load_v_regs = [&](int k0g) {
#pragma unroll
    for (int p = 0; p < 4; ++p) {
      int idx = tid + p * 256;
      int r = idx >> 3, c8 = (idx & 7) * 8;
      vreg[p] = *(const u16x8*)(vb + (size_t)(k0g + r) * 64 + c8);
    }
  };
  auto store_v_trans = [&](int buf) {
#pragma unroll
    for (int p = 0; p < 4; ++p) {
      int idx = tid + p * 256;
      int r = idx >> 3, c8 = (idx & 7) * 8;
#pragma unroll
      for (int i = 0; i < 8; ++i) lsV[buf][(c8 + i) * 136 + r] = vreg[p][i];
    }
  };

  // Q fragments for this wave's 16 rows (two 16x32 bf16 chunks)
  bf16x16 qf[2];
  {
    int qrow = m0 + wv * 16 + l16;
#pragma unroll
    for (int kc = 0; kc < 2; ++kc) {
      u16x8 lo = *(const u16x8*)(qb + (size_t)qrow * 64 + kc * 32 + half * 8);
      u16x8 hi = *(const u16x8*)(qb + (size_t)qrow * 64 + kc * 32 + 16 + half * 8);
      qf[kc] = make_frag(lo, hi);
    }
  }

  f32x8 of[4];
  float mrun[8], lrun[8];
#pragma unroll
  for (int nt = 0; nt < 4; ++nt)
#pragma unroll
    for (int e = 0; e < 8; ++e) of[nt][e] = 0.f;
#pragma unroll
  for (int j = 0; j < 8; ++j) { mrun[j] = -1e30f; lrun[j] = 0.f; }

  // prologue: stage tile 0
  issue_k_async(0, 0);
  load_v_regs(0);
  store_v_trans(0);
  wait_asynccnt0();
  __syncthreads();

  int pb = 0;
  for (int kt = 0; kt < 16; ++kt) {
    const bool haveNext = (kt + 1 < 16);
    if (haveNext) {                       // overlap next tile with compute
      issue_k_async(pb ^ 1, (kt + 1) * 128);
      load_v_regs((kt + 1) * 128);
    }

    // --- S = Qs @ K^T : 8 tiles of 16x16 (scale pre-folded into Q) ---
    f32x8 sc[8];
#pragma unroll
    for (int nt = 0; nt < 8; ++nt) {
      f32x8 z;
#pragma unroll
      for (int e = 0; e < 8; ++e) z[e] = 0.f;
      int row = nt * 16 + l16;
      {
        u16x8 lo = *(const u16x8*)&lsK[pb][row * 72 + half * 8];
        u16x8 hi = *(const u16x8*)&lsK[pb][row * 72 + 16 + half * 8];
        z = wmma_bf16(qf[0], make_frag(lo, hi), z);
      }
      {
        u16x8 lo = *(const u16x8*)&lsK[pb][row * 72 + 32 + half * 8];
        u16x8 hi = *(const u16x8*)&lsK[pb][row * 72 + 32 + 16 + half * 8];
        z = wmma_bf16(qf[1], make_frag(lo, hi), z);
      }
      sc[nt] = z;
    }

    // --- row max (reduce across 16-lane group) ---
    float mnew[8], resc[8];
#pragma unroll
    for (int j = 0; j < 8; ++j) {
      float m = sc[0][j];
#pragma unroll
      for (int nt = 1; nt < 8; ++nt) m = fmaxf(m, sc[nt][j]);
#pragma unroll
      for (int off = 1; off < 16; off <<= 1)
        m = fmaxf(m, __shfl_xor(m, off, 32));
      mnew[j] = fmaxf(mrun[j], m);
      resc[j] = __expf(mrun[j] - mnew[j]);
    }

    // --- P = exp(S - mnew), row sums, running stats ---
#pragma unroll
    for (int nt = 0; nt < 8; ++nt)
#pragma unroll
      for (int j = 0; j < 8; ++j)
        sc[nt][j] = __expf(sc[nt][j] - mnew[j]);

#pragma unroll
    for (int j = 0; j < 8; ++j) {
      float s = 0.f;
#pragma unroll
      for (int nt = 0; nt < 8; ++nt) s += sc[nt][j];
#pragma unroll
      for (int off = 1; off < 16; off <<= 1) s += __shfl_xor(s, off, 32);
      lrun[j] = lrun[j] * resc[j] + s;
      mrun[j] = mnew[j];
    }

    // --- rescale O ---
#pragma unroll
    for (int nt2 = 0; nt2 < 4; ++nt2)
#pragma unroll
      for (int j = 0; j < 8; ++j) of[nt2][j] *= resc[j];

    // --- P -> per-wave LDS as bf16 (C-layout to memory) ---
    const int pbase = wv * 16 * 136;
#pragma unroll
    for (int nt = 0; nt < 8; ++nt)
#pragma unroll
      for (int j = 0; j < 8; ++j)
        lsP[pbase + (half * 8 + j) * 136 + nt * 16 + l16] = f32_to_bf16(sc[nt][j]);

    // --- O += P @ V : K dim = 128 in 4 WMMA steps, 4 output column tiles ---
#pragma unroll
    for (int kc = 0; kc < 4; ++kc) {
      u16x8 plo = *(const u16x8*)&lsP[pbase + l16 * 136 + kc * 32 + half * 8];
      u16x8 phi = *(const u16x8*)&lsP[pbase + l16 * 136 + kc * 32 + 16 + half * 8];
      bf16x16 pf = make_frag(plo, phi);
#pragma unroll
      for (int nt2 = 0; nt2 < 4; ++nt2) {
        int rowd = nt2 * 16 + l16;
        u16x8 vlo = *(const u16x8*)&lsV[pb][rowd * 136 + kc * 32 + half * 8];
        u16x8 vhi = *(const u16x8*)&lsV[pb][rowd * 136 + kc * 32 + 16 + half * 8];
        of[nt2] = wmma_bf16(pf, make_frag(vlo, vhi), of[nt2]);
      }
    }

    if (haveNext) {
      store_v_trans(pb ^ 1);
      wait_asynccnt0();
      __syncthreads();
      pb ^= 1;
    }
  }

  // --- normalize and write (B,T,D) f32 ---
  const int b = bh >> 4, h = bh & 15;
#pragma unroll
  for (int nt2 = 0; nt2 < 4; ++nt2)
#pragma unroll
    for (int j = 0; j < 8; ++j) {
      int t   = m0 + wv * 16 + half * 8 + j;
      int col = h * 64 + nt2 * 16 + l16;
      attn_out[((size_t)b * 2048 + t) * 1024 + col] = of[nt2][j] / lrun[j];
    }
}

// ---------------------------------------------------------------------------
extern "C" void kernel_launch(void* const* d_in, const int* /*in_sizes*/,
                              int /*n_in*/, void* d_out, int /*out_size*/,
                              void* d_ws, size_t /*ws_size*/,
                              hipStream_t stream)
{
  const float* x      = (const float*)d_in[0];
  const float* qkv_w  = (const float*)d_in[1];
  const float* qkv_b  = (const float*)d_in[2];
  const float* proj_w = (const float*)d_in[3];
  const float* proj_b = (const float*)d_in[4];
  float* out = (float*)d_out;

  unsigned short* qkv_ws = (unsigned short*)d_ws;                       // 48 MB bf16
  float* attn_ws = (float*)((char*)d_ws + (size_t)3 * 8192 * 1024 * 2); // 32 MB f32

  dim3 gq(3072 / 128, 8192 / 128);
  gemm_xwT<0><<<gq, 256, 0, stream>>>(x, qkv_w, qkv_b, qkv_ws, nullptr,
                                      8192, 3072, 1024);

  dim3 ga(2048 / 128, 64);
  flash_attn<<<ga, 256, 0, stream>>>(qkv_ws, attn_ws);

  dim3 gp(1024 / 128, 8192 / 128);
  gemm_xwT<1><<<gp, 256, 0, stream>>>(attn_ws, proj_w, proj_b, nullptr, out,
                                      8192, 1024, 1024);
}
